// ScaledDotProduct_13967233646854
// MI455X (gfx1250) — matrix operand order repaired and checked by reading, not compiled, not run on hardware
//
#include <hip/hip_runtime.h>
#include <cstdint>

// ---------------------------------------------------------------------------
// Flash-attention for B=4, L=4096, D=64 on MI455X (gfx1250, wave32).
//   scores = (Q K^T) * (1/8) * mask ; P = softmax(scores) ; out = P V
// Online softmax over 32-key chunks; bf16 WMMA 16x16x32 with f32 accumulation.
// Mask (268 MB fp32) is the bandwidth floor; compute runs under it.
// ---------------------------------------------------------------------------

typedef __bf16 bf16;
typedef __attribute__((ext_vector_type(16))) bf16  v16bf;
typedef __attribute__((ext_vector_type(8)))  bf16  v8bf;
typedef __attribute__((ext_vector_type(2)))  bf16  v2bf;
typedef __attribute__((ext_vector_type(8)))  float v8f;
typedef __attribute__((ext_vector_type(4)))  float v4f;

static constexpr int kB    = 4;
static constexpr int kL    = 4096;
static constexpr int kD    = 64;
static constexpr int kWaves = 4;      // waves per workgroup
static constexpr int kQBlk  = 16 * kWaves;  // 64 queries per workgroup
static constexpr int kKC    = 32;     // keys per chunk
static constexpr int kVTS   = 40;     // Vt row stride in bf16 (pad 32->40: 16B-aligned rows, spread LDS banks)

__global__ __launch_bounds__(kWaves * 32)
void fattn_kernel(const float* __restrict__ Qg, const float* __restrict__ Kg,
                  const float* __restrict__ Vg, const float* __restrict__ Mg,
                  float* __restrict__ Og) {
  __shared__ __attribute__((aligned(16))) bf16  Vt[kD * kVTS];          // V chunk, transposed [d][key], bf16
  __shared__ __attribute__((aligned(16))) float Pl[kWaves][16][kKC];    // per-wave P transpose scratch

  const int tid  = threadIdx.x;
  const int lane = tid & 31;
  const int wid  = tid >> 5;
  const int b    = blockIdx.y;
  const int q0   = blockIdx.x * kQBlk + wid * 16;  // first query row of this wave
  const int m    = lane & 15;
  const int half = lane >> 4;            // 0: lanes 0-15, 1: lanes 16-31
  const int kbA  = half ? 16 : 0;        // 16-bit A-fragment K offset (contiguous 16-runs)
  const int kbB  = half ? 16 : 0;        // 16-bit B-fragment K offset (contiguous 16-runs)

  const float scale = 0.125f;            // 1/sqrt(64)

  const float* Qb = Qg + (size_t)b * kL * kD;
  const float* Kb = Kg + (size_t)b * kL * kD;
  const float* Vb = Vg + (size_t)b * kL * kD;
  const float* Mb = Mg + (size_t)b * kL * kL;

  // ---- Q A-fragments (16x32 bf16, per t-step over d), scale folded in ----
  v16bf qa[2];
  {
    const float* qrow = Qb + (size_t)(q0 + m) * kD;
    #pragma unroll
    for (int t = 0; t < 2; ++t) {
      const int d0 = t * 32 + kbA;
      v4f f0 = *(const v4f*)(qrow + d0);
      v4f f1 = *(const v4f*)(qrow + d0 + 4);
      v4f f2 = *(const v4f*)(qrow + d0 + 8);
      v4f f3 = *(const v4f*)(qrow + d0 + 12);
      #pragma unroll
      for (int i = 0; i < 4; ++i) {
        qa[t][i]      = (bf16)(f0[i] * scale);
        qa[t][i + 4]  = (bf16)(f1[i] * scale);
        qa[t][i + 8]  = (bf16)(f2[i] * scale);
        qa[t][i + 12] = (bf16)(f3[i] * scale);
      }
    }
  }

  // ---- online-softmax state ----
  v8f o0 = {}, o1 = {}, o2 = {}, o3 = {};      // O accumulators, C-layout, n-chunks 0..3
  float rowM[8], rowS[8];
  #pragma unroll
  for (int r = 0; r < 8; ++r) { rowM[r] = -1e30f; rowS[r] = 0.0f; }

  // V staging mapping: thread -> key pair (2*kp, 2*kp+1), 8 d-values
  const int kp    = tid & 15;
  const int dbase = (tid >> 4) * 8;

  for (int kc = 0; kc < kL; kc += kKC) {
    __syncthreads();  // previous chunk's Vt readers done before overwrite

    // ---- cooperative stage: V chunk (32x64 fp32) -> Vt (bf16, [d][key]) ----
    {
      const float* v0 = Vb + (size_t)(kc + 2 * kp)     * kD + dbase;
      const float* v1 = Vb + (size_t)(kc + 2 * kp + 1) * kD + dbase;
      v4f a0 = *(const v4f*)(v0);
      v4f a1 = *(const v4f*)(v0 + 4);
      v4f b0 = *(const v4f*)(v1);
      v4f b1 = *(const v4f*)(v1 + 4);
      #pragma unroll
      for (int j = 0; j < 8; ++j) {
        float fa = (j < 4) ? a0[j] : a1[j - 4];
        float fb = (j < 4) ? b0[j] : b1[j - 4];
        v2bf pr;
        pr[0] = (bf16)fa;
        pr[1] = (bf16)fb;
        *(v2bf*)(&Vt[(dbase + j) * kVTS + 2 * kp]) = pr;
      }
    }

    // ---- prefetch next mask tile (global_prefetch_b8) ----
    if (kc + kKC < kL) {
      __builtin_prefetch(Mb + (size_t)(q0 + m) * kL + (kc + kKC) + half * 16, 0, 1);
    }

    // ---- S = (scale*Q) K^T : two 16x16 f32 tiles (keys 0-15, 16-31) ----
    v8f s0 = {}, s1 = {};
    #pragma unroll
    for (int t = 0; t < 2; ++t) {
      // B-fragment: 16 contiguous d-values along the key row
      const float* k0row = Kb + (size_t)(kc + m) * kD + t * 32 + kbB;
      const float* k1row = Kb + (size_t)(kc + 16 + m) * kD + t * 32 + kbB;
      v16bf kf0, kf1;
      {
        v4f f0 = *(const v4f*)(k0row);
        v4f f1 = *(const v4f*)(k0row + 4);
        v4f f2 = *(const v4f*)(k0row + 8);
        v4f f3 = *(const v4f*)(k0row + 12);
        v4f g0 = *(const v4f*)(k1row);
        v4f g1 = *(const v4f*)(k1row + 4);
        v4f g2 = *(const v4f*)(k1row + 8);
        v4f g3 = *(const v4f*)(k1row + 12);
        #pragma unroll
        for (int i = 0; i < 4; ++i) {
          kf0[i]      = (bf16)f0[i];  kf1[i]      = (bf16)g0[i];
          kf0[i + 4]  = (bf16)f1[i];  kf1[i + 4]  = (bf16)g1[i];
          kf0[i + 8]  = (bf16)f2[i];  kf1[i + 8]  = (bf16)g2[i];
          kf0[i + 12] = (bf16)f3[i];  kf1[i + 12] = (bf16)g3[i];
        }
      }
      s0 = __builtin_amdgcn_wmma_f32_16x16x32_bf16(false, qa[t], false, kf0,
                                                   (short)0, s0, false, false);
      s1 = __builtin_amdgcn_wmma_f32_16x16x32_bf16(false, qa[t], false, kf1,
                                                   (short)0, s1, false, false);
    }

    // ---- multiplicative mask, online softmax, P -> LDS (transpose scratch) ----
    #pragma unroll
    for (int r = 0; r < 8; ++r) {
      const float* mrow = Mb + (size_t)(q0 + r + 8 * half) * kL + kc + m;
      s0[r] *= mrow[0];
      s1[r] *= mrow[16];
    }
    #pragma unroll
    for (int r = 0; r < 8; ++r) {
      float mx = fmaxf(s0[r], s1[r]);
      mx = fmaxf(mx, __shfl_xor(mx, 1, 32));
      mx = fmaxf(mx, __shfl_xor(mx, 2, 32));
      mx = fmaxf(mx, __shfl_xor(mx, 4, 32));
      mx = fmaxf(mx, __shfl_xor(mx, 8, 32));
      const float mn = fmaxf(rowM[r], mx);
      const float cf = __expf(rowM[r] - mn);
      rowM[r] = mn;
      s0[r] = __expf(s0[r] - mn);
      s1[r] = __expf(s1[r] - mn);
      float rs = s0[r] + s1[r];
      rs += __shfl_xor(rs, 1, 32);
      rs += __shfl_xor(rs, 2, 32);
      rs += __shfl_xor(rs, 4, 32);
      rs += __shfl_xor(rs, 8, 32);
      rowS[r] = rowS[r] * cf + rs;
      o0[r] *= cf; o1[r] *= cf; o2[r] *= cf; o3[r] *= cf;
      Pl[wid][r + 8 * half][m]      = s0[r];
      Pl[wid][r + 8 * half][m + 16] = s1[r];
    }

    __syncthreads();  // Vt staged; P transpose scratch written

    // ---- P A-fragment (16x32) from LDS scratch ----
    v16bf pa;
    {
      const float* prow = &Pl[wid][m][0];
      v4f f0 = *(const v4f*)(prow + kbA);
      v4f f1 = *(const v4f*)(prow + kbA + 4);
      v4f f2 = *(const v4f*)(prow + kbA + 8);
      v4f f3 = *(const v4f*)(prow + kbA + 12);
      #pragma unroll
      for (int i = 0; i < 4; ++i) {
        pa[i]      = (bf16)f0[i];
        pa[i + 4]  = (bf16)f1[i];
        pa[i + 8]  = (bf16)f2[i];
        pa[i + 12] = (bf16)f3[i];
      }
    }

    // ---- O += P V : 4 n-chunks of 16 d-columns, K = 32 keys ----
    #pragma unroll
    for (int nc = 0; nc < 4; ++nc) {
      const bf16* vrow = &Vt[(nc * 16 + m) * kVTS + kbB];
      v8bf lo = *(const v8bf*)(vrow);
      v8bf hi = *(const v8bf*)(vrow + 8);
      v16bf vf;
      #pragma unroll
      for (int i = 0; i < 8; ++i) { vf[i] = lo[i]; vf[i + 8] = hi[i]; }
      if (nc == 0) o0 = __builtin_amdgcn_wmma_f32_16x16x32_bf16(false, pa, false, vf, (short)0, o0, false, false);
      if (nc == 1) o1 = __builtin_amdgcn_wmma_f32_16x16x32_bf16(false, pa, false, vf, (short)0, o1, false, false);
      if (nc == 2) o2 = __builtin_amdgcn_wmma_f32_16x16x32_bf16(false, pa, false, vf, (short)0, o2, false, false);
      if (nc == 3) o3 = __builtin_amdgcn_wmma_f32_16x16x32_bf16(false, pa, false, vf, (short)0, o3, false, false);
    }
  }

  // ---- normalize and write out (C-layout: lane holds column n, rows r+8*half) ----
  float* Ob = Og + (size_t)b * kL * kD;
  #pragma unroll
  for (int r = 0; r < 8; ++r) {
    const float inv = 1.0f / rowS[r];
    const size_t row = (size_t)(q0 + r + 8 * half) * kD;
    Ob[row +  0 + m] = o0[r] * inv;
    Ob[row + 16 + m] = o1[r] * inv;
    Ob[row + 32 + m] = o2[r] * inv;
    Ob[row + 48 + m] = o3[r] * inv;
  }
}

extern "C" void kernel_launch(void* const* d_in, const int* in_sizes, int n_in,
                              void* d_out, int out_size, void* d_ws, size_t ws_size,
                              hipStream_t stream) {
  (void)in_sizes; (void)n_in; (void)d_ws; (void)ws_size; (void)out_size;
  const float* q = (const float*)d_in[0];
  const float* k = (const float*)d_in[1];
  const float* v = (const float*)d_in[2];
  const float* m = (const float*)d_in[3];
  // d_in[4] = d_k (64), constant-folded into the kernel.
  dim3 grid(kL / kQBlk, kB);
  fattn_kernel<<<grid, kWaves * 32, 0, stream>>>(q, k, v, m, (float*)d_out);
}
